// MLP_10505490006276
// MI455X (gfx1250) — matrix operand order repaired and checked by reading, compiled, not run on hardware
//
#include <hip/hip_runtime.h>
#include <hip/hip_bf16.h>

// Problem sizes (fixed by the reference)
#define B_N   16384
#define D_N   2048
#define E_N   10
#define H1_N  512
#define H2_N  64

// Tiling
#define BLOCK_M   64                 // rows per GEMM block (4 x 16-row WMMA subtiles)
#define TILES_CAP 272                // >= sum ceil(cnt_e/64) <= 256+10
#define PERM_CAP  (TILES_CAP * 64)   // 17408 padded rows

typedef __attribute__((ext_vector_type(16))) __bf16 v16bf;
typedef __attribute__((ext_vector_type(8)))  float  v8f;

// ---- workspace byte offsets (all 64B aligned) ----
#define WS_COUNTS   0            // 16 ints
#define WS_CURSORS  64           // 16 ints
#define WS_POFF     128          // 16 ints (E+1 used)
#define WS_TILEEXP  256          // TILES_CAP ints
#define WS_PERM     4096         // PERM_CAP ints           (ends 73728)
#define WS_W2B      81920        // 16*4*32*16 bf16 = 64KB  (ends 147456)
#define WS_W1B      147456       // 10*64*32*32*16 bf16 = 20971520 B (ends 21118976)
#define WS_XG       21118976     // PERM_CAP*2048 bf16 = 71303168 B  (~92.4 MB total)

__device__ __forceinline__ unsigned short f2bf(float f) {
    unsigned u = __float_as_uint(f);
    u = (u + 0x7FFFu + ((u >> 16) & 1u)) >> 16;   // round-to-nearest-even
    return (unsigned short)u;
}
__device__ __forceinline__ unsigned pack2(float a, float b) {
    return (unsigned)f2bf(a) | ((unsigned)f2bf(b) << 16);
}

// ---------------- routing kernels ----------------
__global__ void k_init(int* wsi) {
    int idx = blockIdx.x * blockDim.x + threadIdx.x;
    if (idx < 16) { wsi[WS_COUNTS/4 + idx] = 0; wsi[WS_CURSORS/4 + idx] = 0; }
    if (idx < TILES_CAP) wsi[WS_TILEEXP/4 + idx] = 0;
    if (idx < PERM_CAP)  wsi[WS_PERM/4 + idx] = -1;
}

__global__ void k_hist(const int* __restrict__ y, int* counts) {
    int i = blockIdx.x * 256 + threadIdx.x;
    if (i < B_N) atomicAdd(&counts[y[i]], 1);
}

__global__ void k_build(int* wsi) {
    if (threadIdx.x == 0 && blockIdx.x == 0) {
        int po = 0;
        for (int e = 0; e < E_N; ++e) {
            wsi[WS_POFF/4 + e] = po;
            int nt = (wsi[WS_COUNTS/4 + e] + 63) / 64;
            for (int c = 0; c < nt; ++c) wsi[WS_TILEEXP/4 + po/64 + c] = e;
            po += nt * 64;
        }
        wsi[WS_POFF/4 + E_N] = po;
    }
}

__global__ void k_scatter(const int* __restrict__ y, int* wsi) {
    int i = blockIdx.x * 256 + threadIdx.x;
    if (i < B_N) {
        int e   = y[i];
        int pos = wsi[WS_POFF/4 + e] + atomicAdd(&wsi[WS_CURSORS/4 + e], 1);
        wsi[WS_PERM/4 + pos] = i;
    }
}

// ---------------- fp32 -> bf16 pre-swizzle of W1, W2 into WMMA B layout ----------------
// B-operand layout (16-bit B 32x16): lane = ((K>>3)&1)*16 + N%16, elem = (K&7) + 8*((K>>4)&1).
// Flat bf16 index W1b: (((e*64+kc)*32 + nt)*32 + lane)*16 + elem
__global__ void k_convert(const float* __restrict__ W1, const float* __restrict__ W2,
                          unsigned* __restrict__ w1b, unsigned* __restrict__ w2b) {
    long long u = (long long)blockIdx.x * 256 + threadIdx.x;
    const long long NW1 = (long long)E_N * 64 * 32 * 32 * 8;   // 5,242,880 uints
    if (u < NW1) {
        int e = (int)(u >> 19);
        int kc = (int)(u >> 13) & 63, nt = (int)(u >> 8) & 31;
        int lane = (int)(u >> 3) & 31, j = (int)u & 7;
        int kk = (j < 4) ? ((lane >> 4) * 8 + 2 * j) : (16 + (lane >> 4) * 8 + 2 * (j - 4));
        int k  = kc * 32 + kk;
        int n  = nt * 16 + (lane & 15);
        const float* p = W1 + ((long long)e * D_N + k) * H1_N + n;
        w1b[u] = pack2(p[0], p[H1_N]);           // pair differs in K -> stride H1
    } else {
        long long v = u - NW1;
        if (v < 16 * 4 * 32 * 8) {
            int kc = (int)(v >> 10), nt = (int)(v >> 8) & 3;
            int lane = (int)(v >> 3) & 31, j = (int)v & 7;
            int kk = (j < 4) ? ((lane >> 4) * 8 + 2 * j) : (16 + (lane >> 4) * 8 + 2 * (j - 4));
            int k  = kc * 32 + kk;
            int n  = nt * 16 + (lane & 15);
            w2b[v] = pack2(W2[k * H2_N + n], W2[(k + 1) * H2_N + n]);
        }
    }
}

// ---------------- gather x rows (via perm) -> bf16 in WMMA A layout ----------------
// A-operand layout (16-bit A 16x32): lane = M + 16*((K>>3)&1), elem = (K&7) + 8*((K>>4)&1).
// Flat bf16 index xg: ((g*64 + kc)*32 + lane)*16 + elem , g = 16-row subtile index.
__global__ void k_gather(const float* __restrict__ x, const int* __restrict__ perm,
                         unsigned* __restrict__ xg) {
    long long u = (long long)blockIdx.x * 256 + threadIdx.x;
    if (u >= (long long)PERM_CAP * 1024) return;
    int g = (int)(u >> 14);
    int kc = (int)(u >> 8) & 63, lane = (int)(u >> 3) & 31, j = (int)u & 7;
    int m = lane & 15;
    int s = perm[g * 16 + m];
    unsigned val = 0u;
    if (s >= 0) {
        int base = (lane >> 4) * 8;
        int kk0  = (j < 4) ? (base + 2 * j) : (16 + base + 2 * (j - 4));
        const float* xp = x + (long long)s * D_N + kc * 32 + kk0;
        val = pack2(xp[0], xp[1]);
    }
    __builtin_nontemporal_store(val, xg + u);   // single-use stream: keep L2 for W1b
}

// ---------------- main fused kernel: routed GEMM1 + head (GEMM2 + 64->2 + softmax) ----------------
// 512 threads = 16 wave32. Layer-1: each wave = 2 M-subtiles x 4 N-subtiles (8 acc tiles,
// 64 acc VGPRs) with explicit ping-pong double buffering of A/B operand registers.
__launch_bounds__(512)
__global__ void k_gemm(const __bf16* __restrict__ xg, const __bf16* __restrict__ w1b,
                       const __bf16* __restrict__ w2b,
                       const int* __restrict__ perm, const int* __restrict__ tileExp,
                       const float* __restrict__ b1, const float* __restrict__ b2,
                       const float* __restrict__ W3, const float* __restrict__ b3,
                       float* __restrict__ out) {
    __shared__ __align__(32) unsigned short hS[32768];  // 64 KB, reused twice

    const int t    = blockIdx.x;
    const int e    = tileExp[t];
    const int tid  = threadIdx.x;
    const int w    = tid >> 5, lane = tid & 31;
    const int wm   = w & 1;          // M half: subtiles {2wm, 2wm+1}
    const int wn   = w >> 1;         // 0..7: N-subtile quad {4wn .. 4wn+3}
    const int gB   = t * 4 + wm * 2; // first global 16-row subtile of this wave

    const v16bf* __restrict__ Ab = (const v16bf*)xg;   // 32 B granules
    const v16bf* __restrict__ Bb = (const v16bf*)w1b;

    const v8f zero = {0.f,0.f,0.f,0.f,0.f,0.f,0.f,0.f};
    v8f acc[2][4];
    #pragma unroll
    for (int r = 0; r < 2; ++r)
        #pragma unroll
        for (int c = 0; c < 4; ++c) acc[r][c] = zero;

    auto loadA = [&](v16bf* dst, int kc) {
        #pragma unroll
        for (int r = 0; r < 2; ++r)
            dst[r] = __builtin_nontemporal_load(
                &Ab[((long long)(gB + r) * 64 + kc) * 32 + lane]);
    };
    auto loadB = [&](v16bf* dst, int kc) {
        long long base = (((long long)e * 64 + kc) * 32 + wn * 4) * 32 + lane;
        #pragma unroll
        for (int c = 0; c < 4; ++c) dst[c] = Bb[base + (long long)c * 32];
    };
    auto domma = [&](const v16bf* A2, const v16bf* B4) {
        #pragma unroll
        for (int r = 0; r < 2; ++r)
            #pragma unroll
            for (int c = 0; c < 4; ++c)
                acc[r][c] = __builtin_amdgcn_wmma_f32_16x16x32_bf16(
                    false, A2[r], false, B4[c], (short)0, acc[r][c], false, false);
    };

    v16bf aP[2], bP[4], aQ[2], bQ[4];
    loadA(aP, 0); loadB(bP, 0);
    for (int kc = 0; kc < 64; kc += 2) {
        loadA(aQ, kc + 1); loadB(bQ, kc + 1);      // prefetch next chunk
        domma(aP, bP);                              // compute current
        if (kc + 2 < 64) { loadA(aP, kc + 2); loadB(bP, kc + 2); }
        domma(aQ, bQ);
    }

    // ---- epilogue 1: bias + ReLU, store h as bf16 into LDS in A layout for GEMM2 ----
    #pragma unroll
    for (int r = 0; r < 2; ++r) {
        int msT = wm * 2 + r;
        #pragma unroll
        for (int c = 0; c < 4; ++c) {
            int nt = wn * 4 + c;
            int n  = nt * 16 + (lane & 15);          // global column 0..511
            float bias = b1[e * H1_N + n];
            int kc2   = nt >> 1;
            int elem2 = (lane & 7) + (nt & 1) * 8;
            int l2b   = ((lane >> 3) & 1) * 16;
            #pragma unroll
            for (int j = 0; j < 8; ++j) {
                int m = j + (lane >> 4) * 8;         // row within subtile (C layout)
                float v = acc[r][c][j] + bias;
                v = v > 0.f ? v : 0.f;
                hS[((msT * 16 + kc2) * 32 + (m + l2b)) * 16 + elem2] = f2bf(v);
            }
        }
    }
    __syncthreads();

    // ---- layer-2 GEMM: h(64x512) @ W2(512x64); 16 waves x 1 tile each ----
    const int ms  = w >> 2;     // 0..3
    const int nt2 = w & 3;      // 0..3
    v8f a2 = zero;
    const v16bf* hA  = (const v16bf*)hS;
    const v16bf* W2B = (const v16bf*)w2b;
    for (int kc2 = 0; kc2 < 16; ++kc2) {
        v16bf a  = hA[(ms * 16 + kc2) * 32 + lane];
        v16bf bq = W2B[(kc2 * 4 + nt2) * 32 + lane];
        a2 = __builtin_amdgcn_wmma_f32_16x16x32_bf16(false, a, false, bq, (short)0, a2, false, false);
    }
    __syncthreads();

    // ---- epilogue 2: bias + ReLU, h2 (64x64 fp32) into reused LDS ----
    float* h2S = (float*)hS;
    #pragma unroll
    for (int j = 0; j < 8; ++j) {
        int m = ms * 16 + j + (lane >> 4) * 8;
        int n = nt2 * 16 + (lane & 15);
        float v = a2[j] + b2[n];
        v = v > 0.f ? v : 0.f;
        h2S[m * 64 + n] = v;
    }
    __syncthreads();

    // ---- layer-3 (64->2) + softmax, scatter to out via perm ----
    if (tid < 64) {
        int m = tid;
        int s = perm[t * 64 + m];
        if (s >= 0) {
            float l0 = b3[0], l1 = b3[1];
            #pragma unroll 8
            for (int k = 0; k < 64; ++k) {
                float hv = h2S[m * 64 + k];
                l0 += hv * W3[k * 2 + 0];
                l1 += hv * W3[k * 2 + 1];
            }
            float mx = fmaxf(l0, l1);
            float e0 = __expf(l0 - mx), e1 = __expf(l1 - mx);
            float inv = 1.f / (e0 + e1);
            out[(long long)s * 2 + 0] = e0 * inv;
            out[(long long)s * 2 + 1] = e1 * inv;
        }
    }
}

extern "C" void kernel_launch(void* const* d_in, const int* in_sizes, int n_in,
                              void* d_out, int out_size, void* d_ws, size_t ws_size,
                              hipStream_t stream) {
    (void)in_sizes; (void)n_in; (void)out_size; (void)ws_size;
    const float* x  = (const float*)d_in[0];
    const int*   y  = (const int*)  d_in[1];
    const float* W1 = (const float*)d_in[2];
    const float* b1 = (const float*)d_in[3];
    const float* W2 = (const float*)d_in[4];
    const float* b2 = (const float*)d_in[5];
    const float* W3 = (const float*)d_in[6];
    const float* b3 = (const float*)d_in[7];
    float* out = (float*)d_out;

    char* ws  = (char*)d_ws;
    int*  wsi = (int*)d_ws;
    unsigned* w1b = (unsigned*)(ws + WS_W1B);
    unsigned* w2b = (unsigned*)(ws + WS_W2B);
    unsigned* xg  = (unsigned*)(ws + WS_XG);

    k_init<<<(PERM_CAP + 255) / 256, 256, 0, stream>>>(wsi);
    k_hist<<<(B_N + 255) / 256, 256, 0, stream>>>(y, wsi + WS_COUNTS / 4);
    k_build<<<1, 32, 0, stream>>>(wsi);
    k_scatter<<<(B_N + 255) / 256, 256, 0, stream>>>(y, wsi);

    long long convTot = (long long)E_N * 64 * 32 * 32 * 8 + 16 * 4 * 32 * 8;
    k_convert<<<(int)((convTot + 255) / 256), 256, 0, stream>>>(W1, W2, w1b, w2b);

    long long gTot = (long long)PERM_CAP * 1024;
    k_gather<<<(int)((gTot + 255) / 256), 256, 0, stream>>>(x, wsi + WS_PERM / 4, xg);

    k_gemm<<<TILES_CAP, 512, 0, stream>>>((const __bf16*)xg, (const __bf16*)w1b,
                                          (const __bf16*)w2b,
                                          wsi + WS_PERM / 4, wsi + WS_TILEEXP / 4,
                                          b1, b2, W3, b3, out);
}
